// TritonDynamicAttention_60155311948052
// MI455X (gfx1250) — matrix operand order repaired and checked by reading, not compiled, not run on hardware
//
#include <hip/hip_runtime.h>

// ---------------------------------------------------------------------------
// Block-sparse causal attention for MI455X (gfx1250, wave32, WMMA).
//   Z=1, H=16, S=2048, D=64, block 32x32.
// Kernel 0: one-shot f32->f16 convert of K (row-major) and V (transposed).
// Kernel 1: pool mask -> block activity bytes (lower-triangular only).
// Kernel 2: flash attention, f16 WMMA (16x16x32), f32 accumulation,
//           async global->LDS staging, SGPR bitmask block skipping.
// Workspace layout (needs ~8.5 MB):
//   [0, 64KB)            block mask bytes
//   [64KB, 64KB+4MB)     K in f16, [H][S][D]
//   [64KB+4MB, +8MB)     V in f16 transposed, [H][D][S]
// ---------------------------------------------------------------------------

typedef __attribute__((ext_vector_type(16))) _Float16 v16h;
typedef __attribute__((ext_vector_type(8)))  _Float16 v8h;
typedef __attribute__((ext_vector_type(4)))  _Float16 v4h;
typedef __attribute__((ext_vector_type(8)))  float    v8f;
typedef int v4i __attribute__((vector_size(16)));

#define ATTN_S   2048
#define ATTN_D   64
#define ATTN_H   16
#define ATTN_NB  64   // S / 32

#if __has_builtin(__builtin_amdgcn_global_load_async_to_lds_b128) && \
    __has_builtin(__builtin_amdgcn_s_wait_asynccnt)
#define ATTN_ASYNC_LDS 1
#else
#define ATTN_ASYNC_LDS 0
#endif

__device__ inline v8f wmma_f16(v16h a, v16h b, v8f c) {
  // D = A(16x32 f16) * B(32x16 f16) + C(16x16 f32)
  return __builtin_amdgcn_wmma_f32_16x16x32_f16(
      /*neg_a=*/false, a, /*neg_b=*/false, b,
      /*c_mod=*/(short)0, c, /*reuse_a=*/false, /*reuse_b=*/false);
}

// 16-byte global -> LDS copy; async (ASYNCcnt-tracked) when available.
__device__ inline void cp16_g2l(const _Float16* __restrict__ g,
                                _Float16* __restrict__ l) {
#if ATTN_ASYNC_LDS
  __builtin_amdgcn_global_load_async_to_lds_b128(
      (__attribute__((address_space(1))) v4i*)(__UINTPTR_TYPE__)g,
      (__attribute__((address_space(3))) v4i*)l, /*offset=*/0, /*cpol=*/0);
#else
  *(v8h*)l = *(const v8h*)g;
#endif
}

// ---------------------------------------------------------------------------
// Kernel 0: K/V precision convert.  One WG per (head, 64-key tile).
// K: f32 [H][S][D] -> f16 same layout (coalesced in/out).
// V: f32 [H][S][D] -> f16 [H][D][S] (transpose bounced through LDS).
// ---------------------------------------------------------------------------
__global__ __launch_bounds__(256) void cvt_kernel(
    const float* __restrict__ K, const float* __restrict__ V,
    _Float16* __restrict__ Kh, _Float16* __restrict__ Vt) {
  __shared__ alignas(32) _Float16 tlds[64 * 64];  // V tile transposed [d][key]
  const int hd = blockIdx.x >> 5;   // 16 heads
  const int kb = blockIdx.x & 31;   // 32 key tiles of 64
  const int tid = threadIdx.x;
  const int kbase = kb * 64;

  for (int e = tid; e < 1024; e += 256) {
    const int key = e >> 4;          // 0..63
    const int d4  = (e & 15) * 4;    // 0..60
    const size_t g = ((size_t)(hd * ATTN_S + kbase + key)) * ATTN_D + d4;
    float4 kv = *(const float4*)(K + g);
    v4h kk = {(_Float16)kv.x, (_Float16)kv.y, (_Float16)kv.z, (_Float16)kv.w};
    *(v4h*)(Kh + g) = kk;
    float4 vv = *(const float4*)(V + g);
    tlds[(d4 + 0) * 64 + key] = (_Float16)vv.x;
    tlds[(d4 + 1) * 64 + key] = (_Float16)vv.y;
    tlds[(d4 + 2) * 64 + key] = (_Float16)vv.z;
    tlds[(d4 + 3) * 64 + key] = (_Float16)vv.w;
  }
  __syncthreads();
  {
    const int d  = tid >> 2;          // 0..63
    const int ko = (tid & 3) * 16;    // 0,16,32,48
    v8h a = *(const v8h*)&tlds[d * 64 + ko];
    v8h b = *(const v8h*)&tlds[d * 64 + ko + 8];
    _Float16* op = Vt + ((size_t)(hd * ATTN_D + d)) * ATTN_S + kbase + ko;
    *(v8h*)(op + 0) = a;
    *(v8h*)(op + 8) = b;
  }
}

// ---------------------------------------------------------------------------
// Kernel 1: block mask pooling.  One wave32 per 32x32 int32 block.
// Upper-triangular blocks (bj > bi) are dead under causal masking -> skip read.
// ---------------------------------------------------------------------------
__global__ __launch_bounds__(256) void blockmask_kernel(
    const int* __restrict__ M, unsigned char* __restrict__ BM) {
  const int wid  = (blockIdx.x * 256 + threadIdx.x) >> 5;  // global wave id
  const int lane = threadIdx.x & 31;
  const int hd = wid >> 12;
  const int bi = (wid >> 6) & 63;
  const int bj = wid & 63;

  unsigned char out = 0;
  if (bj <= bi) {
    const int* base = M + ((size_t)(hd * ATTN_S + bi * 32)) * ATTN_S + bj * 32;
    int acc = 0;
    #pragma unroll
    for (int it = 0; it < 8; ++it) {
      const int row = it * 4 + (lane >> 3);
      const int4 x = *(const int4*)(base + (size_t)row * ATTN_S + (lane & 7) * 4);
      acc |= x.x | x.y | x.z | x.w;
    }
    out = (__ballot(acc != 0) != 0ull) ? 1 : 0;
  }
  if (lane == 0) BM[wid] = out;
}

// ---------------------------------------------------------------------------
// Kernel 2: flash attention.  blockDim = 64 (2 waves); wave w owns 16 query
// rows.  One workgroup per (head, 32-row query block).
// ---------------------------------------------------------------------------
__global__ __launch_bounds__(64) void fattn_kernel(
    const float* __restrict__ Q, const _Float16* __restrict__ Kh,
    const _Float16* __restrict__ Vt, const unsigned char* __restrict__ BM,
    float* __restrict__ Out) {
  __shared__ alignas(32) _Float16 Klds[32 * 64];     // K tile [key][d] f16
  __shared__ alignas(32) _Float16 Vtlds[64 * 32];    // V tile [d][key] f16
  __shared__ alignas(32) _Float16 Plds[2][16 * 32];  // per-wave P scratch

  const int hd   = blockIdx.x >> 6;   // head
  const int ib   = blockIdx.x & 63;   // query block (32 rows)
  const int tid  = threadIdx.x;
  const int wave = tid >> 5;
  const int lane = tid & 31;
  const int lh   = lane >> 4;         // 16-lane half index
  const int ln   = lane & 15;

  const int qbase = ib * 32 + wave * 16;      // this wave's first query row
  const float*    Qg  = Q  + (size_t)hd * ATTN_S * ATTN_D;
  const _Float16* Khg = Kh + (size_t)hd * ATTN_S * ATTN_D;
  const _Float16* Vtg = Vt + (size_t)hd * ATTN_D * ATTN_S;

  // ---- Pack this row's block-activity into a wave-uniform 64-bit bitmask:
  // lane l tests bytes l and l+32; ballot results land in SGPRs, so the
  // per-iteration skip test is pure SALU (no memory, no loadcnt wait).
  unsigned long long bmask;
  {
    const unsigned char* bmr = BM + (size_t)(hd * ATTN_NB + ib) * ATTN_NB;
    const int b0 = bmr[lane];
    const int b1 = bmr[lane + 32];
    const unsigned long long lo = __ballot(b0 != 0) & 0xffffffffull;
    const unsigned long long hi = __ballot(b1 != 0) & 0xffffffffull;
    bmask = lo | (hi << 32);
  }

  // ---- Load Q straight into WMMA A-layout (f16), one operand per 32-d step.
  // A k order used here (not ISA table): lane = row (ln), per-lane K chunk of 16 contiguous
  // halves at lh*16 (v16h elements [0..15]), matching the B operand.
  v16h aq[2];
  {
    const float* qp = Qg + (size_t)(qbase + ln) * ATTN_D;
    #pragma unroll
    for (int s = 0; s < 2; ++s) {
      const int d0 = s * 32 + lh * 16;
      float4 c0 = *(const float4*)(qp + d0 + 0);
      float4 c1 = *(const float4*)(qp + d0 + 4);
      float4 c2 = *(const float4*)(qp + d0 + 8);
      float4 c3 = *(const float4*)(qp + d0 + 12);
      v16h a;
      a[0]  = (_Float16)c0.x; a[1]  = (_Float16)c0.y;
      a[2]  = (_Float16)c0.z; a[3]  = (_Float16)c0.w;
      a[4]  = (_Float16)c1.x; a[5]  = (_Float16)c1.y;
      a[6]  = (_Float16)c1.z; a[7]  = (_Float16)c1.w;
      a[8]  = (_Float16)c2.x; a[9]  = (_Float16)c2.y;
      a[10] = (_Float16)c2.z; a[11] = (_Float16)c2.w;
      a[12] = (_Float16)c3.x; a[13] = (_Float16)c3.y;
      a[14] = (_Float16)c3.z; a[15] = (_Float16)c3.w;
      aq[s] = a;
    }
  }

  // All-ones B operand: row-sum of P via one extra WMMA (replaces shuffles).
  v16h ones;
  #pragma unroll
  for (int e = 0; e < 16; ++e) ones[e] = (_Float16)1.0f;

  // ---- Flash-attention running state (row r of C/D tile = element r).
  float mrow[8], lrow[8];
  v8f acc[4];
  {
    v8f z = {};
    #pragma unroll
    for (int c = 0; c < 4; ++c) acc[c] = z;
    #pragma unroll
    for (int r = 0; r < 8; ++r) { mrow[r] = -1e30f; lrow[r] = 0.0f; }
  }

  for (int jb = 0; jb <= ib; ++jb) {
    if (!((bmask >> jb) & 1ull)) continue;   // scalar-only skip

    __syncthreads();  // previous iteration's LDS reads complete

    // ---- Stage K/V tiles (pure f16 copy; async to LDS when available).
    // K tile: 2048 contiguous halves.  V tile: 64 rows of 32 halves,
    // row stride S (pre-transposed), one row per thread.
    {
      const _Float16* kt = Khg + (size_t)(jb * 32) * ATTN_D;
      const _Float16* vt = Vtg + (size_t)tid * ATTN_S + jb * 32;
      #pragma unroll
      for (int c = 0; c < 4; ++c) {
        const int kc = (tid + 64 * c) * 8;
        cp16_g2l(kt + kc, &Klds[kc]);
        cp16_g2l(vt + c * 8, &Vtlds[tid * 32 + c * 8]);
      }
    }
#if ATTN_ASYNC_LDS
    __builtin_amdgcn_s_wait_asynccnt(0);
#endif
    __syncthreads();

    // ---- Scores: S = Q K^T.  B layout: lane = key column (ln), K-dim half
    // selected by lh, 16 contiguous halves along d per lane.
    v8f zero = {};
    v16h b00 = *(const v16h*)&Klds[(0 * 16 + ln) * 64 + 0 * 32 + lh * 16];
    v16h b01 = *(const v16h*)&Klds[(0 * 16 + ln) * 64 + 1 * 32 + lh * 16];
    v16h b10 = *(const v16h*)&Klds[(1 * 16 + ln) * 64 + 0 * 32 + lh * 16];
    v16h b11 = *(const v16h*)&Klds[(1 * 16 + ln) * 64 + 1 * 32 + lh * 16];
    v8f s0 = wmma_f16(aq[0], b00, zero);
    s0     = wmma_f16(aq[1], b01, s0);
    v8f s1 = wmma_f16(aq[0], b10, zero);
    s1     = wmma_f16(aq[1], b11, s1);

    // ---- Element-level causal mask (diagonal block only).
    if (jb == ib) {
      const int kg0 = jb * 32 + ln;
      const int kg1 = kg0 + 16;
      #pragma unroll
      for (int r = 0; r < 8; ++r) {
        const int qg = qbase + r + 8 * lh;
        if (kg0 > qg) s0[r] = -1e30f;
        if (kg1 > qg) s1[r] = -1e30f;
      }
    }

    // ---- Online softmax max: row r's 16 columns live in one VGPR across a
    // 16-lane half -> xor-shuffle masks 1..8 reduce within each half.
    float mnew[8], sc[8];
    #pragma unroll
    for (int r = 0; r < 8; ++r) {
      float mx = fmaxf(s0[r], s1[r]);
      mx = fmaxf(mx, __shfl_xor(mx, 1, 32));
      mx = fmaxf(mx, __shfl_xor(mx, 2, 32));
      mx = fmaxf(mx, __shfl_xor(mx, 4, 32));
      mx = fmaxf(mx, __shfl_xor(mx, 8, 32));
      mnew[r] = fmaxf(mrow[r], mx);
    }
    #pragma unroll
    for (int r = 0; r < 8; ++r) {
      s0[r] = __expf(s0[r] - mnew[r]);
      s1[r] = __expf(s1[r] - mnew[r]);
      sc[r] = __expf(mrow[r] - mnew[r]);
      mrow[r] = mnew[r];
    }
    #pragma unroll
    for (int c = 0; c < 4; ++c)
      #pragma unroll
      for (int r = 0; r < 8; ++r) acc[c][r] *= sc[r];

    // ---- Re-layout P (C/D layout -> A layout) through per-wave LDS scratch.
    _Float16* Pw = &Plds[wave][0];
    #pragma unroll
    for (int r = 0; r < 8; ++r) {
      Pw[(r + 8 * lh) * 32 + ln]      = (_Float16)s0[r];
      Pw[(r + 8 * lh) * 32 + 16 + ln] = (_Float16)s1[r];
    }
    v8h pa0 = *(const v8h*)&Pw[ln * 32 + lh * 16];
    v8h pa1 = *(const v8h*)&Pw[ln * 32 + lh * 16 + 8];
    v16h pA;
    #pragma unroll
    for (int e = 0; e < 8; ++e) { pA[e] = pa0[e]; pA[e + 8] = pa1[e]; }

    // ---- Row sum of P with one WMMA: D[q][*] = sum_k P[q][k]; lands in the
    // same per-row (element r, half lh) mapping as lrow.
    v8f rsum = wmma_f16(pA, ones, zero);
    #pragma unroll
    for (int r = 0; r < 8; ++r) lrow[r] = lrow[r] * sc[r] + rsum[r];

    // ---- PV: acc += P(16x32) * V(32x64); one K=32 step per 16-col tile.
    #pragma unroll
    for (int c = 0; c < 4; ++c) {
      v16h bv = *(const v16h*)&Vtlds[(c * 16 + ln) * 32 + lh * 16];
      acc[c] = wmma_f16(pA, bv, acc[c]);
    }
  }

  // ---- Epilogue: normalize; rows with no active blocks emit zeros.
  float rinv[8];
  #pragma unroll
  for (int r = 0; r < 8; ++r) rinv[r] = (lrow[r] > 0.0f) ? (1.0f / lrow[r]) : 0.0f;
  float* outp = Out + (size_t)hd * ATTN_S * ATTN_D;
  #pragma unroll
  for (int c = 0; c < 4; ++c)
    #pragma unroll
    for (int r = 0; r < 8; ++r)
      outp[(size_t)(qbase + r + 8 * lh) * ATTN_D + c * 16 + ln] = acc[c][r] * rinv[r];
}

// ---------------------------------------------------------------------------
extern "C" void kernel_launch(void* const* d_in, const int* in_sizes, int n_in,
                              void* d_out, int out_size, void* d_ws, size_t ws_size,
                              hipStream_t stream) {
  (void)in_sizes; (void)n_in; (void)out_size; (void)ws_size;
  const float* Q = (const float*)d_in[0];
  const float* K = (const float*)d_in[1];
  const float* V = (const float*)d_in[2];
  const int*   M = (const int*)d_in[3];
  float* Out = (float*)d_out;

  unsigned char* BM  = (unsigned char*)d_ws;                      // 64 KB
  _Float16* KhW = (_Float16*)((char*)d_ws + (64u << 10));          // 4 MB
  _Float16* VtW = (_Float16*)((char*)d_ws + (64u << 10) + (4u << 20)); // 4 MB

  // K/V f16 convert (+V transpose): 16 heads * 32 key tiles.
  cvt_kernel<<<ATTN_H * 32, 256, 0, stream>>>(K, V, KhW, VtW);
  // 16 heads * 64 * 64 blocks = 65536 waves, 8 waves per 256-thread WG.
  blockmask_kernel<<<8192, 256, 0, stream>>>(M, BM);
  // One WG per (head, 32-row query block): 16 * 64 WGs of 64 threads.
  fattn_kernel<<<ATTN_H * ATTN_NB, 64, 0, stream>>>(Q, KhW, VtW, BM, Out);
}
